// sLSTM_33311766348189
// MI455X (gfx1250) — matrix operand order repaired and checked by reading, not compiled, not run on hardware
//
#include <hip/hip_runtime.h>

// sLSTM: L=2, B=64, T=1024, D=H=512, 4H=2048.
#define L_  2
#define B_  64
#define T_  1024
#define D_  512
#define H_  512
#define G4_ 2048
#define K_  512   // D_ == H_, both GEMMs share K

typedef __attribute__((ext_vector_type(16))) __bf16         v16bf;
typedef __attribute__((ext_vector_type(8)))  float          v8f;
typedef __attribute__((ext_vector_type(8)))  unsigned short us8;
typedef __attribute__((ext_vector_type(4)))  unsigned       v4u;
typedef __attribute__((ext_vector_type(8)))  int            v8i_;
typedef __attribute__((ext_vector_type(4)))  int            v4i_;

__device__ __forceinline__ unsigned short f2bf(float f) {
  unsigned u = __builtin_bit_cast(unsigned, f);
  u += 0x7FFFu + ((u >> 16) & 1u);             // round-to-nearest-even
  return (unsigned short)(u >> 16);
}
__device__ __forceinline__ float bf2f(unsigned short h) {
  unsigned u = ((unsigned)h) << 16;
  return __builtin_bit_cast(float, u);
}
__device__ __forceinline__ float sigmoid_f(float x) {
  return 1.0f / (1.0f + __expf(-x));
}
__device__ __forceinline__ float tanh_f(float x) {
  x = fminf(fmaxf(x, -15.0f), 15.0f);
  float e = __expf(2.0f * x);
  return (e - 1.0f) / (e + 1.0f);
}

// ---------------------------------------------------------------------------
// CDNA5 async global->LDS copy (ASYNCcnt-tracked, bypasses VGPRs).
// VDST VGPR holds the LDS byte address; per ISA §10.2 the low 32 bits of a
// generic LDS pointer ARE the LDS offset.
// ---------------------------------------------------------------------------
__device__ __forceinline__ void async_copy16(unsigned short* lds_dst,
                                             const unsigned short* gsrc) {
  unsigned ldsa = (unsigned)(unsigned long long)lds_dst;
  unsigned long long ga = (unsigned long long)gsrc;
  asm volatile("global_load_async_to_lds_b128 %0, %1, off"
               :: "v"(ldsa), "v"(ga)
               : "memory");
}
__device__ __forceinline__ void wait_async0() {
  asm volatile("s_wait_asynccnt 0x0" ::: "memory");
}

// ---------------------------------------------------------------------------
// Device-wide split barrier (generation based; self-resetting; counters are
// re-zeroed by prep_kernel at the start of every kernel_launch call).
// ---------------------------------------------------------------------------
__device__ __forceinline__ void grid_sync(unsigned* cnt, unsigned* gen, unsigned nwg) {
  __threadfence();
  __syncthreads();
  if (threadIdx.x == 0) {
    unsigned g = __hip_atomic_load(gen, __ATOMIC_ACQUIRE, __HIP_MEMORY_SCOPE_AGENT);
    unsigned a = __hip_atomic_fetch_add(cnt, 1u, __ATOMIC_ACQ_REL, __HIP_MEMORY_SCOPE_AGENT);
    if (a == nwg - 1u) {
      __hip_atomic_store(cnt, 0u, __ATOMIC_RELAXED, __HIP_MEMORY_SCOPE_AGENT);
      __hip_atomic_fetch_add(gen, 1u, __ATOMIC_ACQ_REL, __HIP_MEMORY_SCOPE_AGENT);
    } else {
      while (__hip_atomic_load(gen, __ATOMIC_ACQUIRE, __HIP_MEMORY_SCOPE_AGENT) == g) {
        __builtin_amdgcn_s_sleep(2);
      }
    }
  }
  __syncthreads();
}

// ---------------------------------------------------------------------------
// Prep: fold biases, convert x / Wx / Wh to bf16, zero sync counters.
// ---------------------------------------------------------------------------
__global__ void prep_kernel(const float* __restrict__ x,  const float* __restrict__ Wx,
                            const float* __restrict__ bx, const float* __restrict__ Wh,
                            const float* __restrict__ bh, const float* __restrict__ bext,
                            unsigned short* __restrict__ xbf,
                            unsigned short* __restrict__ Wxbf,
                            unsigned short* __restrict__ Whbf,
                            float* __restrict__ bias, unsigned* __restrict__ sync) {
  long i = (long)blockIdx.x * blockDim.x + threadIdx.x;
  if (i < (long)B_ * T_ * D_) xbf[i] = f2bf(x[i]);
  if (i < (long)L_ * G4_ * K_) { Wxbf[i] = f2bf(Wx[i]); Whbf[i] = f2bf(Wh[i]); }
  if (i < L_ * G4_) bias[i] = bx[i] + bh[i] + bext[i];
  if (i < 32) sync[i] = 0u;
}

// ---------------------------------------------------------------------------
// Input projection GEMM: xg[r, n] = A[r, :] @ Wx[n, :] + bias[n], bf16 out.
// M = T*B (row r = t*B + b), N = 4H, K = 512. 128x128x32 tiles, 8 waves.
// Double-buffered LDS staging via CDNA5 async global->LDS copies:
//   issue async tile k+1 -> WMMA on tile k -> s_wait_asynccnt 0 -> barrier.
// ---------------------------------------------------------------------------
__global__ __launch_bounds__(256, 2)
void xgemm_kernel(const unsigned short* __restrict__ Asrc, int amode,
                  const unsigned short* __restrict__ Wxbf_l,
                  const float* __restrict__ bias_l,
                  unsigned short* __restrict__ xg) {
  __shared__ __align__(32) unsigned short At[2][8][32][16];   // 16 KB
  __shared__ __align__(32) unsigned short Bt[2][8][32][16];   // 16 KB

  const int tid  = threadIdx.x;
  const int lane = tid & 31;
  const int wave = tid >> 5;
  const int half = lane >> 4;
  const int l16  = lane & 15;
  const int m0   = blockIdx.x * 128;
  const int n0   = blockIdx.y * 128;

  // Staging identity: thread = (tile smt, lane) lane-row of both tiles.
  const int smt  = wave;
  const int arow = m0 + smt * 16 + l16;
  const unsigned short* arow_ptr;
  if (amode == 0) {           // layer 0: x is (B,T,D); logical row r = t*B + b
    int t = arow >> 6;        // / B_
    int b = arow & 63;
    arow_ptr = Asrc + ((long)b * T_ + t) * D_;
  } else {                    // layer 1: hs is (T,B,H) row-major in r
    arow_ptr = Asrc + (long)arow * K_;
  }
  const unsigned short* brow_ptr = Wxbf_l + (long)(n0 + smt * 16 + l16) * K_;

  auto stage = [&](int kk, int b) {
    int kb  = kk + half * 8;                      // A: two 8-elem runs {kb, kb+16}
    async_copy16(&At[b][smt][lane][0], arow_ptr + kb);
    async_copy16(&At[b][smt][lane][8], arow_ptr + kb + 16);
    int kb2 = kk + half * 8;                      // B: two 8-elem runs {kb2, kb2+16}
    async_copy16(&Bt[b][smt][lane][0], brow_ptr + kb2);
    async_copy16(&Bt[b][smt][lane][8], brow_ptr + kb2 + 16);
  };

  v8f zero;
#pragma unroll
  for (int q = 0; q < 8; ++q) zero[q] = 0.0f;

  v8f acc[2][4];
#pragma unroll
  for (int mi = 0; mi < 2; ++mi)
#pragma unroll
    for (int ni = 0; ni < 4; ++ni) acc[mi][ni] = zero;

  const int mtb = (wave & 3) * 2;   // this wave's 2 M-tiles
  const int ntb = (wave >> 2) * 4;  // this wave's 4 N-tiles

  stage(0, 0);
  wait_async0();
  __syncthreads();

  int buf = 0;
  for (int kk = 0; kk < K_; kk += 32) {
    if (kk + 32 < K_) stage(kk + 32, buf ^ 1);    // async prefetch next tile

    v16bf bfrag[4];
#pragma unroll
    for (int ni = 0; ni < 4; ++ni)
      bfrag[ni] = *(const v16bf*)&Bt[buf][ntb + ni][lane][0];
#pragma unroll
    for (int mi = 0; mi < 2; ++mi) {
      v16bf afrag = *(const v16bf*)&At[buf][mtb + mi][lane][0];
#pragma unroll
      for (int ni = 0; ni < 4; ++ni)
        acc[mi][ni] = __builtin_amdgcn_wmma_f32_16x16x32_bf16(
            false, afrag, false, bfrag[ni], (short)0, acc[mi][ni], false, false);
    }

    wait_async0();      // next buffer landed in LDS
    __syncthreads();    // (also orders this buffer's reads vs. next overwrite)
    buf ^= 1;
  }

  // Epilogue: C/D layout — VGPR r: lanes 0-15 => M=r, lanes 16-31 => M=8+r.
#pragma unroll
  for (int mi = 0; mi < 2; ++mi) {
    int mrow = m0 + (mtb + mi) * 16 + half * 8;
#pragma unroll
    for (int ni = 0; ni < 4; ++ni) {
      int col = n0 + (ntb + ni) * 16 + l16;
      float bc = bias_l[col];
#pragma unroll
      for (int r = 0; r < 8; ++r)
        xg[(long)(mrow + r) * G4_ + col] = f2bf(acc[mi][ni][r] + bc);
    }
  }
}

// ---------------------------------------------------------------------------
// Persistent recurrent scan. 32 WGs x 128 threads (4 waves). WG wg owns hidden
// units j in [16wg, 16wg+16). Its 64x512 bf16 Wh slice is DMA'd into LDS by
// the Tensor Data Mover as one 3D tile (4 gates x 16 rows x 512), row-major
// with 16B of TDM pad per 1024B row (260-DWORD row stride -> fragment lanes
// hit banks 0,4,...,60: conflict-free). h lives in global memory (L2-resident,
// 64 KB) in pre-swizzled WMMA A-fragment layout. Wave w = batch rows
// [16w,16w+16); per step, 16 kc x 4 gates = 64 WMMAs, fully unrolled.
// ---------------------------------------------------------------------------
#define BW_ROW 520   // 512 elems + 8 pad elems (4 DWORDs), in ushorts

__global__ __launch_bounds__(128, 1)
void slstm_scan_kernel(const unsigned short* __restrict__ xg,     // (T,B,4H) bf16
                       const unsigned short* __restrict__ Whbf_l, // (4H,K) bf16
                       unsigned short* __restrict__ hA,           // 64x512 A-frag bf16
                       unsigned short* __restrict__ hs_next,      // (T,B,H) bf16 | null
                       float* __restrict__ out,                   // (B,T,H) f32 | null
                       float* __restrict__ hN, float* __restrict__ cN,
                       float* __restrict__ nN,
                       unsigned* __restrict__ cnt, unsigned* __restrict__ gen) {
  __shared__ __align__(32) unsigned short Bw[4 * 16 * BW_ROW];   // 66560 B

  const int tid  = threadIdx.x;
  const int lane = tid & 31;
  const int wave = tid >> 5;            // == M-tile (batch rows 16*wave..)
  const int wg   = blockIdx.x;          // hidden units [16wg, 16wg+16)
  const int half = lane >> 4;
  const int l16  = lane & 15;

  // --- TDM: one descriptor loads the whole Wh slice (gate g, row y, k x):
  //   global addr = base + 2*(x + y*512 + g*512*512), base = Whbf + 16*wg*512.
  if (wave == 0) {
    unsigned long long ga =
        (unsigned long long)(Whbf_l + (unsigned long long)wg * 16u * K_);
    unsigned ldsa = (unsigned)(unsigned long long)(&Bw[0]);
    v4u g0;
    g0[0] = 1u;                                   // count=1 (valid user D#)
    g0[1] = ldsa;                                 // lds_addr
    g0[2] = (unsigned)(ga & 0xFFFFFFFFull);       // global_addr[31:0]
    g0[3] = (unsigned)((ga >> 32) & 0x01FFFFFFull) | (2u << 30);  // [56:32]|type=2
    v8i_ g1;
    g1[0] = (int)0x07D10000;    // data_size=2B | pad_en | pad_int=256dw | pad_amt=4dw
    g1[1] = (int)(512u << 16);  // tensor_dim0 = 512   (bits 79:48, low half)
    g1[2] = (int)(16u << 16);   // tensor_dim1 = 16    (bits 111:80, low half)
    g1[3] = (int)(512u << 16);  // tile_dim0   = 512   (bits 127:112)
    g1[4] = (int)((4u << 16) | 16u);   // tile_dim1=16, tile_dim2=4
    g1[5] = 512;                // tensor_dim0_stride[31:0] = 512
    g1[6] = 0;                  // ^[47:32]=0, tensor_dim1_stride[15:0]=0
    g1[7] = 4;                  // tensor_dim1_stride[47:16] = 262144>>16
    v4i_ g2; g2[0] = 4; g2[1] = 0; g2[2] = 0; g2[3] = 0;   // tensor_dim2=4
    v4i_ g3; g3[0] = 0; g3[1] = 0; g3[2] = 0; g3[3] = 0;
    v8i_ g4;                    // trailing group (zero-filled, 6-arg toolchain form)
#pragma unroll
    for (int q = 0; q < 8; ++q) g4[q] = 0;
    __builtin_amdgcn_tensor_load_to_lds(g0, g1, g2, g3, g4, 0);
    __builtin_amdgcn_s_wait_tensorcnt(0);
  }
  __syncthreads();

  float c_st[8], n_st[8];
#pragma unroll
  for (int r = 0; r < 8; ++r) { c_st[r] = 0.0f; n_st[r] = 0.0f; }

  const int j = wg * 16 + l16;          // this thread's hidden unit (all r)
  // Inverse A-frag mapping for scatter of h at K-index j:
  const int kc_j = j >> 5;
  const int kk_j = j & 31;
  int elem, lhalf;
  if      (kk_j <  8) { lhalf = 0; elem = kk_j;      }
  else if (kk_j < 16) { lhalf = 1; elem = kk_j -  8; }
  else if (kk_j < 24) { lhalf = 0; elem = kk_j -  8; }
  else                { lhalf = 1; elem = kk_j - 16; }

  // B-fragment base for this lane in the row-major TDM tile:
  //   frag (g,kc): col = lane&15, k = kc*32 + (lane>>4)*8 + {0..7, 16..23}
  const int bw_lane = l16 * BW_ROW + half * 8;

  v8f zero;
#pragma unroll
  for (int q = 0; q < 8; ++q) zero[q] = 0.0f;

  for (int t = 0; t < T_; ++t) {
    v8f acc[4];
#pragma unroll
    for (int g = 0; g < 4; ++g) acc[g] = zero;

    if (t > 0) {    // h(t=0) == 0, skip the GEMM entirely
#pragma unroll
      for (int kc = 0; kc < 16; ++kc) {
        v16bf a = *(const v16bf*)(hA + ((wave * 16 + kc) * 512 + lane * 16));
#pragma unroll
        for (int g = 0; g < 4; ++g) {
          const unsigned short* p = &Bw[(g * 16 * BW_ROW) + bw_lane + kc * 32];
          union { us8 h2[2]; v16bf v; } u;
          u.h2[0] = *(const us8*)(p);
          u.h2[1] = *(const us8*)(p + 16);
          acc[g] = __builtin_amdgcn_wmma_f32_16x16x32_bf16(
              false, a, false, u.v, (short)0, acc[g], false, false);
        }
      }
    }

    const unsigned short* xgt = xg + (long)t * B_ * G4_;
#pragma unroll
    for (int r = 0; r < 8; ++r) {
      int b = wave * 16 + half * 8 + r;                 // C/D row mapping
      const unsigned short* xr = xgt + (long)b * G4_ + wg * 16 + l16;
      float gi = acc[0][r] + bf2f(xr[0 * H_]);
      float gf = acc[1][r] + bf2f(xr[1 * H_]);
      float go = acc[2][r] + bf2f(xr[2 * H_]);
      float gu = acc[3][r] + bf2f(xr[3 * H_]);
      float iv = sigmoid_f(gi);
      float fv = sigmoid_f(gf);
      float ov = sigmoid_f(go);
      float uv = tanh_f(gu);
      float cv = fv * c_st[r] + iv * uv;
      float nv = fminf(fmaxf(fv * n_st[r] + iv, 1e-6f), 1e6f);
      float hv = ov * tanh_f(cv / nv);
      c_st[r] = cv; n_st[r] = nv;

      unsigned short hb = f2bf(hv);
      int mt = b >> 4, mr = b & 15;
      hA[(mt * 16 + kc_j) * 512 + (mr + 16 * lhalf) * 16 + elem] = hb;
      if (hs_next) hs_next[((long)t * B_ + b) * H_ + j] = hb;
      if (out)     out[((long)b * T_ + t) * H_ + j] = hv;
      if (t == T_ - 1) {
        hN[b * H_ + j] = hv; cN[b * H_ + j] = cv; nN[b * H_ + j] = nv;
      }
      if (t + 1 < T_) __builtin_prefetch(xr + B_ * G4_, 0, 1);  // next step's xg
    }
    grid_sync(cnt, gen, gridDim.x);   // publish h(t) before any WG loads it
  }
}

// ---------------------------------------------------------------------------
extern "C" void kernel_launch(void* const* d_in, const int* in_sizes, int n_in,
                              void* d_out, int out_size, void* d_ws, size_t ws_size,
                              hipStream_t stream) {
  (void)in_sizes; (void)n_in; (void)out_size; (void)ws_size;
  const float* x    = (const float*)d_in[0];
  const float* Wx   = (const float*)d_in[1];
  const float* bx   = (const float*)d_in[2];
  const float* Wh   = (const float*)d_in[3];
  const float* bh   = (const float*)d_in[4];
  const float* bext = (const float*)d_in[5];

  char* ws = (char*)d_ws;
  size_t off = 0;
  auto alloc = [&](size_t bytes) {
    char* p = ws + off;
    off += (bytes + 255) & ~(size_t)255;
    return p;
  };
  unsigned short* xg   = (unsigned short*)alloc((size_t)T_ * B_ * G4_ * 2); // 256 MB
  unsigned short* xbf  = (unsigned short*)alloc((size_t)B_ * T_ * D_ * 2);  //  64 MB
  unsigned short* hsbf = (unsigned short*)alloc((size_t)T_ * B_ * H_ * 2);  //  64 MB
  unsigned short* Wxbf = (unsigned short*)alloc((size_t)L_ * G4_ * K_ * 2); //   4 MB
  unsigned short* Whbf = (unsigned short*)alloc((size_t)L_ * G4_ * K_ * 2); //   4 MB
  float*          bias = (float*)alloc((size_t)L_ * G4_ * 4);
  unsigned short* hA   = (unsigned short*)alloc((size_t)B_ * K_ * 2);       //  64 KB
  unsigned*       sync = (unsigned*)alloc(256);

  float* outp = (float*)d_out;                       // (B,T,H)
  float* hN   = outp + (size_t)B_ * T_ * H_;         // (L,B,H)
  float* cN   = hN + (size_t)L_ * B_ * H_;
  float* nN   = cN + (size_t)L_ * B_ * H_;

  {
    long total = (long)B_ * T_ * D_;
    int blocks = (int)((total + 255) / 256);
    prep_kernel<<<blocks, 256, 0, stream>>>(x, Wx, bx, Wh, bh, bext,
                                            xbf, Wxbf, Whbf, bias, sync);
  }

  for (int l = 0; l < L_; ++l) {
    const unsigned short* Asrc = (l == 0) ? xbf : hsbf;
    xgemm_kernel<<<dim3(512, 16), 256, 0, stream>>>(
        Asrc, l, Wxbf + (size_t)l * G4_ * K_, bias + (size_t)l * G4_, xg);
    slstm_scan_kernel<<<32, 128, 0, stream>>>(
        xg, Whbf + (size_t)l * G4_ * K_, hA,
        (l == 0) ? hsbf : (unsigned short*)nullptr,
        (l == 1) ? outp : (float*)nullptr,
        hN + (size_t)l * B_ * H_, cN + (size_t)l * B_ * H_,
        nN + (size_t)l * B_ * H_,
        sync, sync + 16);
  }
}
